// UnifiedGNN_55413668053095
// MI455X (gfx1250) — compile-verified, matrix operand order checked by reading, not run on hardware
//
#include <hip/hip_runtime.h>
#include <math.h>

typedef __bf16 bf16;
typedef __attribute__((ext_vector_type(16))) __bf16 v16bf;
typedef __attribute__((ext_vector_type(8)))  float  v8f;

#define NP 16384
#define NS 32768
#define NT 49152            // NP + NS
#define FD 256
#define NHEAD 8
#define NRELROWS 114688     // 3*NP + 2*NS
#define ETOT 229376
#define NB 512

// ---------------------------------------------------------------- helpers
__device__ __forceinline__ float gelu_f(float x) {
    return 0.5f * x * (1.0f + erff(x * 0.7071067811865476f));
}
__device__ __forceinline__ int f2ord(float x) {
    int i = __float_as_int(x);
    return i >= 0 ? i : (i ^ 0x7fffffff);
}
__device__ __forceinline__ float ord2f(int o) {
    int i = o >= 0 ? o : (o ^ 0x7fffffff);
    return __int_as_float(i);
}

// Async global -> LDS copy (CDNA5 ASYNCcnt path). LDS byte address is the
// low 32 bits of the generic pointer (LDS aperture keeps wave-relative
// offset in bits [31:0]).
__device__ __forceinline__ void async_copy_b128(const void* lds, const void* g) {
    unsigned ldsOff = (unsigned)(uintptr_t)lds;
    unsigned long long ga = (unsigned long long)(uintptr_t)g;
    asm volatile("global_load_async_to_lds_b128 %0, %1, off"
                 :: "v"(ldsOff), "v"(ga) : "memory");
}
__device__ __forceinline__ void wait_async0() {
    asm volatile("s_wait_asynccnt 0x0" ::: "memory");
}

// ------------------------------------------------ input linear + GELU (K small)
__global__ __launch_bounds__(256)
void in_linear_kernel(const float* __restrict__ x, const float* __restrict__ w,
                      const float* __restrict__ b, float* __restrict__ xf,
                      bf16* __restrict__ xb, int KF, int n) {
    int idx = blockIdx.x * 256 + threadIdx.x;
    if (idx >= n * FD) return;
    int node = idx >> 8, f = idx & 255;
    float s = b[f];
    for (int k = 0; k < KF; ++k) s += x[node * KF + k] * w[k * FD + f];
    float g = gelu_f(s);
    xf[idx] = g;
    xb[idx] = (bf16)g;
}

// ------------------------------------------------ WMMA GEMM: C = act(A[M,256]@W[256,N]+bias)
// 64x64 block tile, 8 waves, each wave one 16x32 tile (2 wmma per K-step).
// A tile staged via async global->LDS; W tile converted f32->bf16 transposed.
__global__ __launch_bounds__(256)
void gemm_bf16_256(const bf16* __restrict__ A, const float* __restrict__ W,
                   const float* __restrict__ bias,
                   float* __restrict__ Cf, bf16* __restrict__ Cb,
                   const float* __restrict__ skipX,
                   const float* __restrict__ skipGate,
                   int N, int applyGelu) {
    __shared__ alignas(64) bf16 As[64][32];
    __shared__ alignas(64) bf16 Wt[64][32];   // transposed: Wt[n][k]
    int tid = threadIdx.x, lane = tid & 31, wave = tid >> 5;
    int wrow = wave & 3, wcol = wave >> 2;
    size_t m0 = (size_t)blockIdx.x * 64;
    int n0 = blockIdx.y * 64;
    int ar = tid >> 2, ac = (tid & 3) * 8;      // A staging: 16B / thread
    int wk = tid >> 3, wn = (tid & 7) * 8;      // W staging: 8 f32 / thread
    v8f c0 = {}, c1 = {};
    for (int kk = 0; kk < 256; kk += 32) {
        __syncthreads();
        async_copy_b128(&As[ar][ac], A + (m0 + ar) * 256 + kk + ac);
        const float* wp = W + (size_t)(kk + wk) * N + n0 + wn;
#pragma unroll
        for (int j = 0; j < 8; ++j) Wt[wn + j][wk] = (bf16)wp[j];
        if (kk + 32 < 256)
            __builtin_prefetch(A + (m0 + ar) * 256 + kk + 32 + ac, 0, 1);
        wait_async0();
        __syncthreads();
        v16bf a  = *(const v16bf*)&As[wrow * 16 + (lane & 15)][(lane >> 4) * 16];
        v16bf b0 = *(const v16bf*)&Wt[wcol * 32 + (lane & 15)][(lane >> 4) * 16];
        v16bf b1 = *(const v16bf*)&Wt[wcol * 32 + 16 + (lane & 15)][(lane >> 4) * 16];
        c0 = __builtin_amdgcn_wmma_f32_16x16x32_bf16(false, a, false, b0,
                                                     (short)0, c0, false, false);
        c1 = __builtin_amdgcn_wmma_f32_16x16x32_bf16(false, a, false, b1,
                                                     (short)0, c1, false, false);
    }
    float gate = 0.f;
    if (skipX) gate = 1.f / (1.f + expf(-skipGate[0]));
    size_t mBase = m0 + wrow * 16 + ((lane >> 4) << 3);
#pragma unroll
    for (int j = 0; j < 2; ++j) {
        v8f c = j ? c1 : c0;
        int col = n0 + wcol * 32 + j * 16 + (lane & 15);
        float bc = bias ? bias[col] : 0.f;
#pragma unroll
        for (int r = 0; r < 8; ++r) {
            size_t off = (mBase + r) * (size_t)N + col;
            float v = c[r] + bc;
            if (skipX) v = gate * v + (1.f - gate) * skipX[off];
            if (applyGelu) v = gelu_f(v);
            if (Cf) Cf[off] = v;
            if (Cb) Cb[off] = (bf16)v;
        }
    }
}

// ------------------------------------------------ per-head 32x32 relation transform
// Y[n, h*32+f] = sum_d X[n, h*32+d] * R[h][d][f]   (K=32 -> one wmma per tile)
__global__ __launch_bounds__(256)
void headmm_kernel(const bf16* __restrict__ X, const float* __restrict__ R,
                   bf16* __restrict__ Y) {
    int h = blockIdx.y;
    __shared__ alignas(64) bf16 As[64][32];
    __shared__ alignas(64) bf16 Bt[32][32];   // Bt[f][d]
    int tid = threadIdx.x, lane = tid & 31, wave = tid >> 5;
    size_t m0 = (size_t)blockIdx.x * 64;
    { int r = tid >> 2, cb = (tid & 3) * 8;
      async_copy_b128(&As[r][cb], X + (m0 + r) * 256 + h * 32 + cb); }
    { int d = tid >> 3, fb = (tid & 7) * 4;
      const float* rp = R + ((size_t)h * 32 + d) * 32 + fb;
#pragma unroll
      for (int j = 0; j < 4; ++j) Bt[fb + j][d] = (bf16)rp[j]; }
    wait_async0();
    __syncthreads();
    int wrow = wave & 3, wcol = wave >> 2;
    v16bf a = *(const v16bf*)&As[wrow * 16 + (lane & 15)][(lane >> 4) * 16];
    v16bf b = *(const v16bf*)&Bt[wcol * 16 + (lane & 15)][(lane >> 4) * 16];
    v8f c = {};
    c = __builtin_amdgcn_wmma_f32_16x16x32_bf16(false, a, false, b,
                                                (short)0, c, false, false);
    size_t mBase = m0 + wrow * 16 + ((lane >> 4) << 3);
    int col = h * 32 + wcol * 16 + (lane & 15);
#pragma unroll
    for (int r = 0; r < 8; ++r) Y[(mBase + r) * 256 + col] = (bf16)c[r];
}

// ------------------------------------------------ attention state init
__global__ __launch_bounds__(256)
void att_init_kernel(float* __restrict__ agg, float* __restrict__ den,
                     int* __restrict__ mx) {
    int idx = blockIdx.x * 256 + threadIdx.x;
    if (idx < NT * FD) agg[idx] = 0.f;
    if (idx < NT * NHEAD) { den[idx] = 0.f; mx[idx] = (int)0x80000000; }
}

// ------------------------------------------------ edge pass 1: logits + segment max
__global__ __launch_bounds__(256)
void edge_logit_kernel(const int* __restrict__ e, int ne,
                       const bf16* __restrict__ krelSlot,
                       const bf16* __restrict__ q, int dstBase,
                       const float* __restrict__ prel8,
                       float* __restrict__ lg, int* __restrict__ mx) {
    int idx = blockIdx.x * 256 + threadIdx.x;
    if (idx >= ne * NHEAD) return;
    int eid = idx >> 3, h = idx & 7;
    int src = e[eid], dst = e[ne + eid];
    const bf16* kr = krelSlot + (size_t)src * 256 + h * 32;
    const bf16* qr = q + (size_t)(dstBase + dst) * 256 + h * 32;
    float s = 0.f;
#pragma unroll
    for (int d = 0; d < 32; ++d) s += (float)kr[d] * (float)qr[d];
    float v = s * prel8[h] * 0.17677669529663687f;   // 1/sqrt(32)
    lg[idx] = v;
    atomicMax(&mx[(dstBase + dst) * NHEAD + h], f2ord(v));
}

// ------------------------------------------------ edge pass 2: exp + scatter-add
__global__ __launch_bounds__(256)
void edge_accum_kernel(const int* __restrict__ e, int ne,
                       const bf16* __restrict__ vrelSlot, int dstBase,
                       const float* __restrict__ lg, const int* __restrict__ mx,
                       float* __restrict__ den, float* __restrict__ agg) {
    int idx = blockIdx.x * 256 + threadIdx.x;
    if (idx >= ne * NHEAD) return;
    int eid = idx >> 3, h = idx & 7;
    int src = e[eid], dst = e[ne + eid];
    int node = dstBase + dst;
    float m = ord2f(mx[node * NHEAD + h]);
    float a = expf(lg[idx] - m);
    atomicAdd(&den[node * NHEAD + h], a);
    const bf16* vr = vrelSlot + (size_t)src * 256 + h * 32;
    float* ag = agg + (size_t)node * 256 + h * 32;
    for (int d = 0; d < 32; ++d) atomicAdd(&ag[d], a * (float)vr[d]);
}

// ------------------------------------------------ normalize + GELU -> bf16
__global__ __launch_bounds__(256)
void finalize_kernel(const float* __restrict__ agg, const float* __restrict__ den,
                     bf16* __restrict__ hbuf) {
    int idx = blockIdx.x * 256 + threadIdx.x;
    if (idx >= NT * FD) return;
    int node = idx >> 8, h = (idx >> 5) & 7;
    float d = den[node * NHEAD + h];
    float x = d > 0.f ? agg[idx] / d : 0.f;
    hbuf[idx] = (bf16)gelu_f(x);
}

// ------------------------------------------------ per-board max pool over pieces
__global__ __launch_bounds__(256)
void pool_kernel(const float* __restrict__ xs, float* __restrict__ out) {
    int idx = blockIdx.x * 256 + threadIdx.x;
    if (idx >= NB * FD) return;
    int b = idx >> 8, f = idx & 255;
    float m = -3.4e38f;
    for (int i = 0; i < 32; ++i)
        m = fmaxf(m, xs[((size_t)b * 32 + i) * 256 + f]);
    out[idx] = m;
}

// ================================================================ launcher
extern "C" void kernel_launch(void* const* d_in, const int* in_sizes, int n_in,
                              void* d_out, int out_size, void* d_ws, size_t ws_size,
                              hipStream_t stream) {
    (void)in_sizes; (void)n_in; (void)out_size; (void)ws_size;
    const float* piece_x  = (const float*)d_in[0];
    const float* square_x = (const float*)d_in[1];
    const int* eptr[5] = {(const int*)d_in[3], (const int*)d_in[4],
                          (const int*)d_in[5], (const int*)d_in[6],
                          (const int*)d_in[7]};
    const float* lin_p_w = (const float*)d_in[8];
    const float* lin_p_b = (const float*)d_in[9];
    const float* lin_s_w = (const float*)d_in[10];
    const float* lin_s_b = (const float*)d_in[11];

    static const int ecnt[5]    = {16384, 16384, 32768, 32768, 131072};
    static const int srcT[5]    = {0, 1, 0, 0, 1};
    static const int dstT[5]    = {1, 0, 0, 0, 1};
    static const int slotOff[5] = {0, 16384, 49152, 65536, 81920};
    static const int lgOff[5]   = {0, 16384, 32768, 65536, 98304};

    // ---- workspace layout (aliasing: agg over k/v, h over q)
    char* w8 = (char*)d_ws;
    size_t o = 0;
    auto take = [&](size_t bytes) -> char* {
        char* p = w8 + o; o += (bytes + 255) & ~(size_t)255; return p;
    };
    float* xsA  = (float*)take((size_t)NT * FD * 4);
    float* xsB  = (float*)take((size_t)NT * FD * 4);
    bf16*  xsb  = (bf16*) take((size_t)NT * FD * 2);
    bf16*  qb   = (bf16*) take((size_t)NT * FD * 2);   // q, later reused as h
    bf16*  kb   = (bf16*) take((size_t)NT * FD * 2);
    bf16*  vb   = (bf16*) take((size_t)NT * FD * 2);
    float* agg  = (float*)kb;                          // alias (k+v == NT*FD*4 B)
    bf16*  krel = (bf16*) take((size_t)NRELROWS * FD * 2);
    bf16*  vrel = (bf16*) take((size_t)NRELROWS * FD * 2);
    float* lg   = (float*)take((size_t)ETOT * NHEAD * 4);
    int*   mx   = (int*)  take((size_t)NT * NHEAD * 4);
    float* den  = (float*)take((size_t)NT * NHEAD * 4);

    // ---- input linears + GELU
    in_linear_kernel<<<(NP * FD) / 256, 256, 0, stream>>>(
        piece_x, lin_p_w, lin_p_b, xsA, xsb, 12, NP);
    in_linear_kernel<<<(NS * FD) / 256, 256, 0, stream>>>(
        square_x, lin_s_w, lin_s_b, xsA + (size_t)NP * FD,
        xsb + (size_t)NP * FD, 13, NS);

    for (int l = 0; l < 2; ++l) {
        int ib = 12 + l * 12;
        const float* wkk  = (const float*)d_in[ib + 0];
        const float* bkk  = (const float*)d_in[ib + 1];
        const float* wqq  = (const float*)d_in[ib + 2];
        const float* bqq  = (const float*)d_in[ib + 3];
        const float* wvv  = (const float*)d_in[ib + 4];
        const float* bvv  = (const float*)d_in[ib + 5];
        const float* woo  = (const float*)d_in[ib + 6];
        const float* boo  = (const float*)d_in[ib + 7];
        const float* skp  = (const float*)d_in[ib + 8];
        const float* arel = (const float*)d_in[ib + 9];
        const float* mrel = (const float*)d_in[ib + 10];
        const float* prel = (const float*)d_in[ib + 11];

        float* xin  = l ? xsB : xsA;
        float* xout = l ? xsA : xsB;

        // Q/K/V projections (WMMA GEMMs)
        for (int t = 0; t < 2; ++t) {
            size_t toff = t ? (size_t)NP * FD : 0;
            int Mt = t ? NS : NP;
            dim3 g(Mt / 64, FD / 64);
            gemm_bf16_256<<<g, 256, 0, stream>>>(xsb + toff, wqq + t * 65536,
                bqq + t * 256, nullptr, qb + toff, nullptr, nullptr, FD, 0);
            gemm_bf16_256<<<g, 256, 0, stream>>>(xsb + toff, wkk + t * 65536,
                bkk + t * 256, nullptr, kb + toff, nullptr, nullptr, FD, 0);
            gemm_bf16_256<<<g, 256, 0, stream>>>(xsb + toff, wvv + t * 65536,
                bvv + t * 256, nullptr, vb + toff, nullptr, nullptr, FD, 0);
        }
        // per-relation per-head transforms hoisted to node level (WMMA)
        for (int r = 0; r < 5; ++r) {
            int Mt = srcT[r] ? NS : NP;
            size_t soff = srcT[r] ? (size_t)NP * FD : 0;
            dim3 g(Mt / 64, NHEAD);
            headmm_kernel<<<g, 256, 0, stream>>>(kb + soff, arel + r * 8192,
                                                 krel + (size_t)slotOff[r] * FD);
            headmm_kernel<<<g, 256, 0, stream>>>(vb + soff, mrel + r * 8192,
                                                 vrel + (size_t)slotOff[r] * FD);
        }
        // segment-softmax attention over all incoming edges per dst node
        att_init_kernel<<<(NT * FD) / 256, 256, 0, stream>>>(agg, den, mx);
        for (int r = 0; r < 5; ++r)
            edge_logit_kernel<<<(ecnt[r] * NHEAD) / 256, 256, 0, stream>>>(
                eptr[r], ecnt[r], krel + (size_t)slotOff[r] * FD, qb,
                dstT[r] ? NP : 0, prel + r * 8, lg + (size_t)lgOff[r] * NHEAD, mx);
        for (int r = 0; r < 5; ++r)
            edge_accum_kernel<<<(ecnt[r] * NHEAD) / 256, 256, 0, stream>>>(
                eptr[r], ecnt[r], vrel + (size_t)slotOff[r] * FD,
                dstT[r] ? NP : 0, lg + (size_t)lgOff[r] * NHEAD, mx, den, agg);
        finalize_kernel<<<(NT * FD) / 256, 256, 0, stream>>>(agg, den, qb);
        // out-linear + gated skip (+ inter-layer GELU), dual f32/bf16 writeback
        for (int t = 0; t < 2; ++t) {
            size_t toff = t ? (size_t)NP * FD : 0;
            int Mt = t ? NS : NP;
            dim3 g(Mt / 64, FD / 64);
            gemm_bf16_256<<<g, 256, 0, stream>>>(qb + toff, woo + t * 65536,
                boo + t * 256, xout + toff, xsb + toff, xin + toff, skp + t,
                FD, l == 0 ? 1 : 0);
        }
    }
    // global max pool over piece nodes per board (layer-2 output lives in xsA)
    pool_kernel<<<(NB * FD) / 256, 256, 0, stream>>>(xsA, (float*)d_out);
}
